// KeyValueAttention_47278999995104
// MI455X (gfx1250) — compile-verified, matrix operand order checked by reading, not run on hardware
//
#include <hip/hip_runtime.h>
#include <hip/hip_bf16.h>

#define NQc 4096
#define NKc 8192
#define DQK 256
#define DA  64
#define Hc  4
#define DOc 256
#define NSPLIT 4
#define KB_PER_SPLIT (NKc / NSPLIT)      // 2048 keys -> 64 tiles of 32

typedef __attribute__((ext_vector_type(16))) __bf16 bf16x16;
typedef __attribute__((ext_vector_type(8)))  __bf16 bf16x8;
typedef __attribute__((ext_vector_type(8)))  float  f32x8;

__device__ __forceinline__ bf16x16 mk16(bf16x8 lo, bf16x8 hi) {
  return __builtin_shufflevector(lo, hi, 0, 1, 2, 3, 4, 5, 6, 7,
                                 8, 9, 10, 11, 12, 13, 14, 15);
}

__device__ __forceinline__ bf16x8 cvt8(const float* p) {
  bf16x8 r;
#pragma unroll
  for (int i = 0; i < 8; ++i) r[i] = (__bf16)p[i];
  return r;
}

__device__ __forceinline__ f32x8 wmma_bf16(bf16x16 a, bf16x16 b, f32x8 c) {
  return __builtin_amdgcn_wmma_f32_16x16x32_bf16(false, a, false, b, (short)0, c,
                                                 false, false);
}

// Async global->LDS copy, 16B per lane, tracked by ASYNCcnt (gfx1250).
__device__ __forceinline__ void async_ld_b128(uint32_t lds_off, const void* gaddr) {
  asm volatile("global_load_async_to_lds_b128 %0, %1, off"
               :: "v"(lds_off), "v"(gaddr) : "memory");
}
#define WAIT_ASYNC_LE_16() asm volatile("s_wait_asynccnt 0x10" ::: "memory")

// ---------------------------------------------------------------------------
// Per-head projection: Out(16x16 tile) = X(16xK) * W[h](Kx64), bf16 result.
// grid = (M/16, H*DA/16), 1 wave per block. transposeOut=1 stores Vt[h][a][M].
// ---------------------------------------------------------------------------
__global__ __launch_bounds__(32) void proj_bf16(const float* __restrict__ X,
                                                const float* __restrict__ W,
                                                __bf16* __restrict__ Out,
                                                int M, int transposeOut) {
  const int lane = threadIdx.x;
  const int m0 = blockIdx.x * 16;
  const int j  = blockIdx.y;          // 0..15
  const int h  = j >> 2;
  const int a0 = (j & 3) * 16;
  const int l16 = lane & 15;
  const int cb  = (lane >> 4) * 8;    // per-half K offset (ISA 16-bit layout)

  const float* Wh = W + (size_t)h * DQK * DA;
  f32x8 acc;
#pragma unroll
  for (int v = 0; v < 8; ++v) acc[v] = 0.f;

  for (int kk = 0; kk < DQK; kk += 32) {
    const float* ar = X + (size_t)(m0 + l16) * DQK + kk;
    bf16x16 A = mk16(cvt8(ar + cb), cvt8(ar + 16 + cb));
    bf16x8 blo, bhi;
#pragma unroll
    for (int i = 0; i < 8; ++i) {
      blo[i] = (__bf16)Wh[(size_t)(kk + cb + i) * DA + a0 + l16];
      bhi[i] = (__bf16)Wh[(size_t)(kk + 16 + cb + i) * DA + a0 + l16];
    }
    acc = wmma_bf16(A, mk16(blo, bhi), acc);
  }

  const int rowOff = (lane >> 4) * 8;
  if (!transposeOut) {
    __bf16* o = Out + ((size_t)h * M + m0 + rowOff) * DA + a0 + l16;
#pragma unroll
    for (int v = 0; v < 8; ++v) o[(size_t)v * DA] = (__bf16)acc[v];
  } else {
    bf16x8 pk;
#pragma unroll
    for (int v = 0; v < 8; ++v) pk[v] = (__bf16)acc[v];
    *(bf16x8*)(Out + ((size_t)h * DA + a0 + l16) * M + m0 + rowOff) = pk;
  }
}

// ---------------------------------------------------------------------------
// gates[h][n] = sigmoid(x_Q[n] . Wg[h] + bg[h]); one wave per row.
// ---------------------------------------------------------------------------
__global__ __launch_bounds__(128) void gates_kernel(const float* __restrict__ xQ,
                                                    const float* __restrict__ Wg,
                                                    const float* __restrict__ bg,
                                                    float* __restrict__ gates) {
  const int wave = threadIdx.x >> 5;
  const int lane = threadIdx.x & 31;
  const int n = blockIdx.x * 4 + wave;
  const float* xr = xQ + (size_t)n * DQK + lane * 8;
  float x[8];
#pragma unroll
  for (int i = 0; i < 8; ++i) x[i] = xr[i];
#pragma unroll
  for (int h = 0; h < Hc; ++h) {
    const float* wr = Wg + h * DQK + lane * 8;
    float d = 0.f;
#pragma unroll
    for (int i = 0; i < 8; ++i) d += x[i] * wr[i];
#pragma unroll
    for (int s = 16; s; s >>= 1) d += __shfl_xor(d, s, 32);
    if (lane == 0) {
      float z = d + bg[h];
      float e = __builtin_amdgcn_exp2f(-z * 1.44269504f);
      gates[h * NQc + n] = __builtin_amdgcn_rcpf(1.f + e);
    }
  }
}

// ---------------------------------------------------------------------------
// Split-K flash attention. grid = (NQ/16, NSPLIT); block = 4 waves (heads).
// K/V tiles double-buffered in LDS via global_load_async_to_lds_b128.
// Writes unnormalized partials pO (16x64), pM/pL (16) per (split, head, qtile).
// ---------------------------------------------------------------------------
__global__ __launch_bounds__(128) void flash_kernel(
    const __bf16* __restrict__ Qh, const __bf16* __restrict__ Kh,
    const __bf16* __restrict__ Vt, const int* __restrict__ mask,
    float* __restrict__ pO, float* __restrict__ pM, float* __restrict__ pL) {
  __shared__ __align__(16) __bf16 ldsK[Hc][2][32 * DA];   // 32KB: 32 keys x 64a
  __shared__ __align__(16) __bf16 ldsV[Hc][2][DA * 32];   // 32KB: 64a x 32 keys
  __shared__ __align__(16) __bf16 ldsP[Hc][16 * 32];      // 4KB

  const int h    = threadIdx.x >> 5;
  const int lane = threadIdx.x & 31;
  const int q0   = blockIdx.x * 16;
  const int split = blockIdx.y;
  const int kb0  = split * KB_PER_SPLIT;
  const int l16  = lane & 15;
  const int cb   = (lane >> 4) * 8;
  const int rowOff = (lane >> 4) * 8;

  // Q fragments: two a-chunks of 32
  const __bf16* qrow = Qh + ((size_t)h * NQc + q0 + l16) * DA;
  bf16x16 qa0 = mk16(*(const bf16x8*)(qrow + cb),      *(const bf16x8*)(qrow + 16 + cb));
  bf16x16 qa1 = mk16(*(const bf16x8*)(qrow + 32 + cb), *(const bf16x8*)(qrow + 48 + cb));

  f32x8 o[4];
  float mi[8], li[8];
#pragma unroll
  for (int v = 0; v < 8; ++v) { mi[v] = -1e30f; li[v] = 0.f; }
#pragma unroll
  for (int nt = 0; nt < 4; ++nt)
#pragma unroll
    for (int v = 0; v < 8; ++v) o[nt][v] = 0.f;

  f32x8 zf;
#pragma unroll
  for (int v = 0; v < 8; ++v) zf[v] = 0.f;

  const float SC  = 0.125f;        // 1/sqrt(64)
  const float L2E = 1.44269504f;

  const int vrow = lane >> 2;            // 0..7
  const int vco  = (lane & 3) * 16;      // byte offset within a 64B V row

  // --- async tile issue: K tile is one contiguous 4KB block; V tile is 64x64B rows
  auto issue_tile = [&](int kb, int buf) {
    const char* gK = (const char*)(Kh + ((size_t)h * NKc + kb) * DA);
    uint32_t lK = (uint32_t)(uintptr_t)&ldsK[h][buf][0];
#pragma unroll
    for (int i = 0; i < 8; ++i) {
      uint32_t off = (uint32_t)(i * 32 + lane) * 16;
      async_ld_b128(lK + off, gK + off);
    }
    uint32_t lV = (uint32_t)(uintptr_t)&ldsV[h][buf][0];
#pragma unroll
    for (int i = 0; i < 8; ++i) {
      int a = i * 8 + vrow;
      async_ld_b128(lV + (uint32_t)(a * 64 + vco),
                    (const char*)Vt + (((size_t)h * DA + a) * NKc + kb) * 2 + vco);
    }
  };

  issue_tile(kb0, 0);   // prologue: tile 0 -> buffer 0

  for (int it = 0; it < KB_PER_SPLIT / 32; ++it) {
    const int kb  = kb0 + it * 32;
    const int buf = it & 1;
    const int nit = (it == KB_PER_SPLIT / 32 - 1) ? 0 : it + 1;  // wrap: harmless reload
    issue_tile(kb0 + nit * 32, (it + 1) & 1);
    WAIT_ASYNC_LE_16();   // current tile's 16 async ops complete (in-order)

    // ---- S = Q * K^T from LDS (two 16x16 key tiles) ----
    const __bf16* kr0 = &ldsK[h][buf][(size_t)l16 * DA];
    const __bf16* kr1 = kr0 + (size_t)16 * DA;
    f32x8 s0 = wmma_bf16(qa0, mk16(*(const bf16x8*)(kr0 + cb),      *(const bf16x8*)(kr0 + 16 + cb)), zf);
    s0       = wmma_bf16(qa1, mk16(*(const bf16x8*)(kr0 + 32 + cb), *(const bf16x8*)(kr0 + 48 + cb)), s0);
    f32x8 s1 = wmma_bf16(qa0, mk16(*(const bf16x8*)(kr1 + cb),      *(const bf16x8*)(kr1 + 16 + cb)), zf);
    s1       = wmma_bf16(qa1, mk16(*(const bf16x8*)(kr1 + 32 + cb), *(const bf16x8*)(kr1 + 48 + cb)), s1);

    // ---- mask + scale (C layout: rows rowOff..rowOff+7, cols l16 / 16+l16) ----
    const int* mrow = mask + (size_t)(q0 + rowOff) * NKc + kb;
    __builtin_prefetch(mrow + 32, 0, 0);
    float p0[8], p1[8];
#pragma unroll
    for (int v = 0; v < 8; ++v) {
      int mv0 = mrow[(size_t)v * NKc + l16];
      int mv1 = mrow[(size_t)v * NKc + 16 + l16];
      p0[v] = mv0 ? s0[v] * SC : -1e30f;
      p1[v] = mv1 ? s1[v] * SC : -1e30f;
    }

    // ---- online softmax ----
#pragma unroll
    for (int v = 0; v < 8; ++v) {
      float mx = fmaxf(p0[v], p1[v]);
      mx = fmaxf(mx, __shfl_xor(mx, 1, 32));
      mx = fmaxf(mx, __shfl_xor(mx, 2, 32));
      mx = fmaxf(mx, __shfl_xor(mx, 4, 32));
      mx = fmaxf(mx, __shfl_xor(mx, 8, 32));
      float mn = fmaxf(mi[v], mx);
      float corr = __builtin_amdgcn_exp2f((mi[v] - mn) * L2E);
      float e0 = __builtin_amdgcn_exp2f((p0[v] - mn) * L2E);
      float e1 = __builtin_amdgcn_exp2f((p1[v] - mn) * L2E);
      float rs = e0 + e1;
      rs += __shfl_xor(rs, 1, 32);
      rs += __shfl_xor(rs, 2, 32);
      rs += __shfl_xor(rs, 4, 32);
      rs += __shfl_xor(rs, 8, 32);
      li[v] = li[v] * corr + rs;
      mi[v] = mn;
#pragma unroll
      for (int nt = 0; nt < 4; ++nt) o[nt][v] *= corr;
      ldsP[h][(rowOff + v) * 32 + l16]      = (__bf16)e0;
      ldsP[h][(rowOff + v) * 32 + 16 + l16] = (__bf16)e1;
    }

    // ---- reload P in A-operand layout ----
    const __bf16* pr = &ldsP[h][l16 * 32];
    bf16x16 pa = mk16(*(const bf16x8*)(pr + cb), *(const bf16x8*)(pr + 16 + cb));

    // ---- O += P * V from LDS ----
#pragma unroll
    for (int nt = 0; nt < 4; ++nt) {
      const __bf16* vr = &ldsV[h][buf][(size_t)(nt * 16 + l16) * 32];
      bf16x16 vb = mk16(*(const bf16x8*)(vr + cb), *(const bf16x8*)(vr + 16 + cb));
      o[nt] = wmma_bf16(pa, vb, o[nt]);
    }
  }

  // ---- write unnormalized partials ----
  const size_t obase = ((size_t)(split * Hc + h) * (NQc / 16) + blockIdx.x) * 1024;
#pragma unroll
  for (int nt = 0; nt < 4; ++nt)
#pragma unroll
    for (int v = 0; v < 8; ++v)
      pO[obase + (size_t)(rowOff + v) * DA + nt * 16 + l16] = o[nt][v];
  if (l16 == 0) {
    const size_t mbase = (size_t)(split * Hc + h) * NQc + q0 + rowOff;
#pragma unroll
    for (int v = 0; v < 8; ++v) { pM[mbase + v] = mi[v]; pL[mbase + v] = li[v]; }
  }
}

// ---------------------------------------------------------------------------
// Merge splits, normalize, apply gates, sum heads -> combined (NQ x 64).
// grid = NQ/16, 128 threads: thread handles (row = t/8, 8 a-values).
// ---------------------------------------------------------------------------
__global__ __launch_bounds__(128) void merge_kernel(
    const float* __restrict__ pO, const float* __restrict__ pM,
    const float* __restrict__ pL, const float* __restrict__ gates,
    float* __restrict__ combined) {
  const int qt = blockIdx.x;
  const int q0 = qt * 16;
  const int r  = threadIdx.x >> 3;
  const int ac = (threadIdx.x & 7) * 8;
  const float L2E = 1.44269504f;

  float acc[8];
#pragma unroll
  for (int i = 0; i < 8; ++i) acc[i] = 0.f;

#pragma unroll
  for (int h = 0; h < Hc; ++h) {
    float mm[NSPLIT];
    float mstar = -1e30f;
#pragma unroll
    for (int s = 0; s < NSPLIT; ++s) {
      mm[s] = pM[(size_t)(s * Hc + h) * NQc + q0 + r];
      mstar = fmaxf(mstar, mm[s]);
    }
    float w[NSPLIT];
    float L = 0.f;
#pragma unroll
    for (int s = 0; s < NSPLIT; ++s) {
      w[s] = __builtin_amdgcn_exp2f((mm[s] - mstar) * L2E);
      L += w[s] * pL[(size_t)(s * Hc + h) * NQc + q0 + r];
    }
    float c = gates[h * NQc + q0 + r] * __builtin_amdgcn_rcpf(L);
#pragma unroll
    for (int s = 0; s < NSPLIT; ++s) {
      float coef = w[s] * c;
      const float* po =
          pO + ((size_t)(s * Hc + h) * (NQc / 16) + qt) * 1024 + (size_t)r * DA + ac;
#pragma unroll
      for (int i = 0; i < 8; ++i) acc[i] += coef * po[i];
    }
  }
#pragma unroll
  for (int i = 0; i < 8; ++i) combined[(size_t)(q0 + r) * DA + ac + i] = acc[i];
}

// ---------------------------------------------------------------------------
// out = combined(NQ x 64) @ Wo(64 x 256) + bo; one wave per 16x16 tile.
// ---------------------------------------------------------------------------
__global__ __launch_bounds__(32) void out_proj(const float* __restrict__ combined,
                                               const float* __restrict__ Wo,
                                               const float* __restrict__ bo,
                                               float* __restrict__ out) {
  const int lane = threadIdx.x;
  const int m0 = blockIdx.x * 16;
  const int n0 = blockIdx.y * 16;
  const int l16 = lane & 15;
  const int cb  = (lane >> 4) * 8;

  f32x8 acc;
#pragma unroll
  for (int v = 0; v < 8; ++v) acc[v] = 0.f;

  for (int kk = 0; kk < DA; kk += 32) {
    const float* ar = combined + (size_t)(m0 + l16) * DA + kk;
    bf16x16 A = mk16(cvt8(ar + cb), cvt8(ar + 16 + cb));
    bf16x8 blo, bhi;
#pragma unroll
    for (int i = 0; i < 8; ++i) {
      blo[i] = (__bf16)Wo[(size_t)(kk + cb + i) * DOc + n0 + l16];
      bhi[i] = (__bf16)Wo[(size_t)(kk + 16 + cb + i) * DOc + n0 + l16];
    }
    acc = wmma_bf16(A, mk16(blo, bhi), acc);
  }

  const int rowOff = (lane >> 4) * 8;
  float b = bo[n0 + l16];
  float* orow = out + (size_t)(m0 + rowOff) * DOc + n0 + l16;
#pragma unroll
  for (int v = 0; v < 8; ++v) orow[(size_t)v * DOc] = acc[v] + b;
}

extern "C" void kernel_launch(void* const* d_in, const int* in_sizes, int n_in,
                              void* d_out, int out_size, void* d_ws, size_t ws_size,
                              hipStream_t stream) {
  const float* xQ   = (const float*)d_in[0];
  const float* xK   = (const float*)d_in[1];
  const int*   mask = (const int*)d_in[2];
  const float* Wq   = (const float*)d_in[3];
  const float* Wk   = (const float*)d_in[4];
  const float* Wv   = (const float*)d_in[5];
  const float* Wg   = (const float*)d_in[6];
  const float* bg   = (const float*)d_in[7];
  const float* Wo   = (const float*)d_in[8];
  const float* bo   = (const float*)d_in[9];
  float* out = (float*)d_out;

  char* ws = (char*)d_ws;
  __bf16* Qh = (__bf16*)(ws);                          // 2 MB
  __bf16* Kh = (__bf16*)(ws + (2ull << 20));           // 4 MB
  __bf16* Vt = (__bf16*)(ws + (6ull << 20));           // 4 MB (transposed V)
  float* gates    = (float*)(ws + (10ull << 20));      // 64 KB
  float* combined = (float*)(ws + (11ull << 20));      // 1 MB
  float* pO = (float*)(ws + (12ull << 20));            // NSPLIT*H*NQ*64 f32 = 16 MB
  float* pM = (float*)(ws + (28ull << 20));            // 256 KB
  float* pL = (float*)(ws + (28ull << 20) + (256ull << 10)); // 256 KB

  dim3 w32(32);
  proj_bf16<<<dim3(NQc / 16, 16), w32, 0, stream>>>(xQ, Wq, Qh, NQc, 0);
  proj_bf16<<<dim3(NKc / 16, 16), w32, 0, stream>>>(xK, Wk, Kh, NKc, 0);
  proj_bf16<<<dim3(NKc / 16, 16), w32, 0, stream>>>(xK, Wv, Vt, NKc, 1);
  gates_kernel<<<dim3(NQc / 4), dim3(128), 0, stream>>>(xQ, Wg, bg, gates);
  flash_kernel<<<dim3(NQc / 16, NSPLIT), dim3(128), 0, stream>>>(Qh, Kh, Vt, mask,
                                                                 pO, pM, pL);
  merge_kernel<<<dim3(NQc / 16), dim3(128), 0, stream>>>(pO, pM, pL, gates, combined);
  out_proj<<<dim3(NQc / 16, DOc / 16), w32, 0, stream>>>(combined, Wo, bo, out);
}
